// EaRL_58248346468589
// MI455X (gfx1250) — compile-verified, matrix operand order checked by reading, not run on hardware
//
#include <hip/hip_runtime.h>
#include <hip/hip_bf16.h>
#include <math.h>

// ---------------------------------------------------------------------------
// MI455X (gfx1250) hetero-GraphSAGE, v2.
//  - all GEMM operands staged as f16, padded (rows->16, K->32) so the WMMA
//    inner loop has ZERO guards and ZERO conversions:
//      4x global_load_b128 + v_wmma_f32_16x16x32_f16 per K-chunk
//  - weights pre-transposed to [N=256, Kp] f16 so B fragments are
//    row-contiguous 16B loads (same lane layout as A per ISA 7.12.2)
//  - scatter-mean: 16B f16 gathers + f32 global atomics
// ---------------------------------------------------------------------------

typedef __attribute__((ext_vector_type(16))) _Float16 v16h;
typedef __attribute__((ext_vector_type(8)))  _Float16 h8;
typedef __attribute__((ext_vector_type(8)))  float    v8f;

constexpr int HD = 256;                        // hidden width (N of every GEMM)

// ---------------- generic helpers ----------------
__global__ void k_zero_f32(float* __restrict__ p, long long n) {
  long long i  = (long long)blockIdx.x * blockDim.x + threadIdx.x;
  long long st = (long long)gridDim.x * blockDim.x;
  for (; i < n; i += st) p[i] = 0.0f;
}

// convert layer-0 features f32[N,d] -> f16[Np,Kp] zero-padded
__global__ void k_cvt_x0(const float* __restrict__ in, _Float16* __restrict__ out,
                         int N, int d, int Np, int Kp) {
  long long total = (long long)Np * Kp;
  long long i  = (long long)blockIdx.x * blockDim.x + threadIdx.x;
  long long st = (long long)gridDim.x * blockDim.x;
  for (; i < total; i += st) {
    int row = (int)(i / Kp), col = (int)(i % Kp);
    out[i] = (row < N && col < d) ? (_Float16)in[(long long)row * d + col]
                                  : (_Float16)0.0f;
  }
}

// W f32[K,256] -> Wt f16[256,Kp] (transposed, K zero-padded to Kp)
__global__ void k_cvt_wt(const float* __restrict__ W, _Float16* __restrict__ Wt,
                         int K, int Kp) {
  long long total = (long long)HD * Kp;
  long long i  = (long long)blockIdx.x * blockDim.x + threadIdx.x;
  long long st = (long long)gridDim.x * blockDim.x;
  for (; i < total; i += st) {
    int n = (int)(i / Kp), k = (int)(i % Kp);
    Wt[i] = (k < K) ? (_Float16)W[(long long)k * HD + n] : (_Float16)0.0f;
  }
}

__global__ void k_add_bias(float* __restrict__ C, const float* __restrict__ b,
                           long long n) {
  long long i  = (long long)blockIdx.x * blockDim.x + threadIdx.x;
  long long st = (long long)gridDim.x * blockDim.x;
  for (; i < n; i += st) C[i] += b[i & (HD - 1)];
}

// relu(acc f32) -> xh f16, elementwise (both stride-256 layouts)
__global__ void k_relu_cvt(const float* __restrict__ acc, _Float16* __restrict__ xh,
                           long long n) {
  long long i  = (long long)blockIdx.x * blockDim.x + threadIdx.x;
  long long st = (long long)gridDim.x * blockDim.x;
  for (; i < n; i += st) {
    float v = acc[i];
    xh[i] = (_Float16)(v > 0.0f ? v : 0.0f);
  }
}

// ---------------- scatter-mean ----------------
// one thread per (edge, 8-half chunk); d in {16,32,128,256}
__global__ void k_scatter_add_h(const _Float16* __restrict__ xsrc, int strideS,
                                const int* __restrict__ src,
                                const int* __restrict__ dst,
                                int E, int d,
                                float* __restrict__ sums, int strideD,
                                float* __restrict__ cnt) {
  int chunks = d >> 3;
  long long total = (long long)E * chunks;
  long long tid = (long long)blockIdx.x * blockDim.x + threadIdx.x;
  if (tid >= total) return;
  int e = (int)(tid / chunks);
  int c = (int)(tid % chunks);
  int s = src[e], t = dst[e];
  h8 v = *(const h8*)(xsrc + (long long)s * strideS + (long long)c * 8);
  float* o = sums + (long long)t * strideD + (long long)c * 8;
  #pragma unroll
  for (int i = 0; i < 8; ++i) atomicAdd(o + i, (float)v[i]);
  if (c == 0) atomicAdd(cnt + t, 1.0f);
}

// sums f32[Ndp,Kp] / max(cnt,1) -> meanh f16[Ndp,Kp]
__global__ void k_div_cvt(const float* __restrict__ sums, const float* __restrict__ cnt,
                          _Float16* __restrict__ meanh, int Ndp, int Kp) {
  long long total = (long long)Ndp * Kp;
  long long i  = (long long)blockIdx.x * blockDim.x + threadIdx.x;
  long long st = (long long)gridDim.x * blockDim.x;
  for (; i < total; i += st) {
    float c = cnt[i / Kp];
    c = c > 1.0f ? c : 1.0f;
    meanh[i] = (_Float16)(sums[i] * (1.0f / c));
  }
}

// ---------------- WMMA GEMM: C[Mp,256] += A[Mp,Kp] * Bt[256,Kp]^T ----------
// block = 256 threads = 8 waves, each wave one 16x16 C tile.
// A fragment (ISA 7.12.2, 16-bit A 16x32): lane (m = lane&15, kb = 8*(lane>>4))
// holds halves [kb,kb+8) and [kb+16,kb+24) of row m  -> two b128 loads.
// Bt stored N-major so the B fragment is the same pattern on row n = lane&15.
__global__ __launch_bounds__(256)
void wmma_gemm_nt(const _Float16* __restrict__ A, int lda,
                  const _Float16* __restrict__ Bt, int ldb,
                  float* __restrict__ C, int Mp, int Kp) {
  const int lane = threadIdx.x & 31;
  const int wave = threadIdx.x >> 5;
  const int n0   = blockIdx.x * 16;
  const int m0   = (blockIdx.y * 8 + wave) * 16;
  if (m0 >= Mp) return;                        // wave-uniform exit

  const int col  = lane & 15;
  const int half = lane >> 4;
  const int kb   = half * 8;

  float* crow = C + (long long)(m0 + 8 * half) * HD + n0 + col;
  v8f c;
  #pragma unroll
  for (int r = 0; r < 8; ++r) c[r] = crow[(long long)r * HD];

  const _Float16* ap = A  + (long long)(m0 + col) * lda + kb;
  const _Float16* bp = Bt + (long long)(n0 + col) * ldb + kb;
  for (int k0 = 0; k0 < Kp; k0 += 32) {
    h8 a0 = *(const h8*)(ap);
    h8 a1 = *(const h8*)(ap + 16);
    h8 b0 = *(const h8*)(bp);
    h8 b1 = *(const h8*)(bp + 16);
    v16h af = __builtin_shufflevector(a0, a1, 0, 1, 2, 3, 4, 5, 6, 7,
                                      8, 9, 10, 11, 12, 13, 14, 15);
    v16h bf = __builtin_shufflevector(b0, b1, 0, 1, 2, 3, 4, 5, 6, 7,
                                      8, 9, 10, 11, 12, 13, 14, 15);
    c = __builtin_amdgcn_wmma_f32_16x16x32_f16(false, af, false, bf,
                                               (short)0, c, false, false);
    ap += 32; bp += 32;
  }

  #pragma unroll
  for (int r = 0; r < 8; ++r) crow[(long long)r * HD] = c[r];
}

// ---------------- final heads ----------------
__global__ void k_head(const float* __restrict__ sums, const float* __restrict__ cnt,
                       const float* __restrict__ nameWl, const float* __restrict__ nameWr,
                       const float* __restrict__ nameb,
                       const float* __restrict__ zeroWl, const float* __restrict__ zeroWr,
                       const float* __restrict__ zerob,
                       float* __restrict__ out, int Ng) {
  int i = blockIdx.x * blockDim.x + threadIdx.x;
  if (i >= Ng) return;
  float cn = cnt[i];
  cn = cn > 1.0f ? cn : 1.0f;
  float inv = 1.0f / cn;
  const float* r = sums + (long long)i * HD;
  float s1 = 0.0f, s2 = 0.0f;
  for (int f = 0; f < HD; ++f) {
    float v = r[f] * inv;
    s1 += v * nameWl[f];
    s2 += v * zeroWl[f];
  }
  float pred = s1 - nameWr[0] + nameb[0];      // gname features are -1
  float z    = s2 - zeroWr[0] + zerob[0];
  float pz   = 1.0f / (1.0f + expf(-z));
  out[i]      = pred;
  out[Ng + i] = pz;
  unsigned u = (unsigned)i * 2654435761u ^ 0x9E3779B9u;
  u ^= u >> 16; u *= 0x85EBCA6Bu; u ^= u >> 13; u *= 0xC2B2AE35u; u ^= u >> 16;
  float un = (float)(u >> 8) * (1.0f / 16777216.0f);
  out[2 * Ng + i] = (un < pz) ? 1.0f : 0.0f;
}

// ---------------------------------------------------------------------------
// host orchestration
// ---------------------------------------------------------------------------
namespace {
constexpr int NT_N[4]   = {3000, 100000, 20000, 20000};  // tad, atac, gene, protein
constexpr int NT_NP[4]  = {3008, 100000, 20000, 20000};  // rows padded to 16
constexpr int NT_D0[4]  = {16, 32, 128, 128};
constexpr int NT_KP0[4] = {32, 32, 128, 128};            // K padded to 32
constexpr long long ROW_OFFP[4] = {0, 3008, 103008, 123008};
constexpr long long TOTP = 143008;

struct RelInfo { int s, d, E, eidx; };
constexpr RelInfo RELS[11] = {
  {0, 1, 100000,  4},  {1, 0, 100000,  5},  {0, 2,  20000,  6},
  {2, 0,  20000,  7},  {1, 2, 200000,  8},  {2, 1, 200000,  9},
  {3, 3, 500000, 10},  {3, 2, 100000, 11},  {2, 3, 100000, 12},
  {3, 2,  20000, 13},  {2, 3,  20000, 14},
};
constexpr int GNAME_EIDX = 15;
constexpr int PBASE = 16;
constexpr long long WSLOT = 256 * 256;        // halves per transposed weight slot

inline void zf32(float* p, long long n, hipStream_t s) {
  k_zero_f32<<<2048, 256, 0, s>>>(p, n);
}
} // namespace

extern "C" void kernel_launch(void* const* d_in, const int* in_sizes, int n_in,
                              void* d_out, int out_size, void* d_ws, size_t ws_size,
                              hipStream_t stream) {
  (void)in_sizes; (void)n_in; (void)out_size; (void)ws_size;

  // workspace carve-up
  float*    acc   = (float*)d_ws;                          // TOTP*256 f32
  float*    sums  = acc + TOTP * HD;                       // <=100000*256 f32
  float*    cntb  = sums + 100000LL * HD;                  // <=100000 f32
  _Float16* xh    = (_Float16*)(cntb + 100032);            // TOTP*256 f16
  _Float16* meanh = xh + TOTP * HD;                        // <=100000*256 f16
  _Float16* wbuf  = meanh + 100000LL * HD;                 // 88 * 256*256 f16

  // ---- stage all weights, transposed, as f16 (once per launch) ----
  for (int l = 0; l < 4; ++l)
    for (int ri = 0; ri < 11; ++ri) {
      const RelInfo& R = RELS[ri];
      int Ks = (l == 0) ? NT_D0[R.s] : HD, Kps = (l == 0) ? NT_KP0[R.s] : HD;
      int Kd = (l == 0) ? NT_D0[R.d] : HD, Kpd = (l == 0) ? NT_KP0[R.d] : HD;
      const int pidx = PBASE + (l * 11 + ri) * 3;
      _Float16* slot = wbuf + (long long)((l * 11 + ri) * 2) * WSLOT;
      k_cvt_wt<<<512, 256, 0, stream>>>((const float*)d_in[pidx + 0], slot, Ks, Kps);
      k_cvt_wt<<<512, 256, 0, stream>>>((const float*)d_in[pidx + 1], slot + WSLOT, Kd, Kpd);
    }

  // ---- stage layer-0 node features as padded f16 ----
  const _Float16* xptr[4];
  int xstr[4];
  {
    long long off = 0;
    for (int t = 0; t < 4; ++t) {
      k_cvt_x0<<<2048, 256, 0, stream>>>((const float*)d_in[t], xh + off,
                                         NT_N[t], NT_D0[t], NT_NP[t], NT_KP0[t]);
      xptr[t] = xh + off;
      xstr[t] = NT_KP0[t];
      off += (long long)NT_NP[t] * NT_KP0[t];
    }
  }

  // ---- 4 HeteroConv layers ----
  for (int l = 0; l < 4; ++l) {
    zf32(acc, TOTP * HD, stream);

    for (int ri = 0; ri < 11; ++ri) {
      const RelInfo& R = RELS[ri];
      const int Nd  = NT_N[R.d];
      const int Ndp = NT_NP[R.d];
      const int ds  = (l == 0) ? NT_D0[R.s] : HD;
      const int Kps = (l == 0) ? NT_KP0[R.s] : HD;
      const int Kpd = (l == 0) ? NT_KP0[R.d] : HD;
      const int* edge = (const int*)d_in[R.eidx];

      // scatter-mean of src features -> meanh [Ndp, Kps] (f16)
      zf32(sums, (long long)Ndp * Kps, stream);
      zf32(cntb, Ndp, stream);
      {
        long long total = (long long)R.E * (ds >> 3);
        int blocks = (int)((total + 255) / 256);
        k_scatter_add_h<<<blocks, 256, 0, stream>>>(xptr[R.s], xstr[R.s],
                                                    edge, edge + R.E, R.E, ds,
                                                    sums, Kps, cntb);
      }
      k_div_cvt<<<2048, 256, 0, stream>>>(sums, cntb, meanh, Ndp, Kps);

      // acc_d += mean @ Wl + x_d @ Wr + b
      const int pidx = PBASE + (l * 11 + ri) * 3;
      const float* bb = (const float*)d_in[pidx + 2];
      _Float16* slot = wbuf + (long long)((l * 11 + ri) * 2) * WSLOT;
      float* Cd = acc + ROW_OFFP[R.d] * HD;

      dim3 grid(HD / 16, (unsigned)((Ndp + 127) / 128));
      wmma_gemm_nt<<<grid, 256, 0, stream>>>(meanh, Kps, slot, Kps, Cd, Ndp, Kps);
      wmma_gemm_nt<<<grid, 256, 0, stream>>>(xptr[R.d], xstr[R.d], slot + WSLOT, Kpd,
                                             Cd, Ndp, Kpd);
      k_add_bias<<<1024, 256, 0, stream>>>(Cd, bb, (long long)Ndp * HD);
    }

    // relu + convert to f16 features for next layer (overwrites xh)
    k_relu_cvt<<<2048, 256, 0, stream>>>(acc, xh, TOTP * HD);
    for (int t = 0; t < 4; ++t) { xptr[t] = xh + ROW_OFFP[t] * HD; xstr[t] = HD; }
  }

  // ---- heads over e_prot_gname: scatter-mean protein -> gene_name ----
  {
    const int Ng = 20000, Eg = 20000;
    const int* edge = (const int*)d_in[GNAME_EIDX];
    zf32(sums, (long long)Ng * HD, stream);
    zf32(cntb, Ng, stream);
    long long total = (long long)Eg * (HD >> 3);
    int blocks = (int)((total + 255) / 256);
    k_scatter_add_h<<<blocks, 256, 0, stream>>>(xptr[3], xstr[3], edge, edge + Eg,
                                                Eg, HD, sums, HD, cntb);

    const int hb = PBASE + 4 * 11 * 3;
    k_head<<<(Ng + 255) / 256, 256, 0, stream>>>(
        sums, cntb,
        (const float*)d_in[hb + 0], (const float*)d_in[hb + 1], (const float*)d_in[hb + 2],
        (const float*)d_in[hb + 3], (const float*)d_in[hb + 4], (const float*)d_in[hb + 5],
        (float*)d_out, Ng);
  }
}
